// GDMultiheadAttention_33002528702963
// MI455X (gfx1250) — matrix operand order repaired and checked by reading, not compiled, not run on hardware
//
#include <hip/hip_runtime.h>

typedef __attribute__((ext_vector_type(16))) __bf16 v16bf;
typedef __attribute__((ext_vector_type(8)))  float  v8f;
typedef __attribute__((ext_vector_type(4)))  unsigned int u32x4;

union Frag { v16bf bf; u32x4 q[2]; };

__device__ __forceinline__ unsigned short f2bf(float x) {
    union { float f; unsigned int u; } v; v.f = x;
    unsigned int r = v.u + 0x7FFFu + ((v.u >> 16) & 1u);
    return (unsigned short)(r >> 16);
}

// ---------------------------------------------------------------------------
// Kernel 1: QKV projection.  C(4096 x 3072) = X(4096 x 1024) * W_in^T + b_in
// Epilogue scatters into q/k/v buffers of shape (BH=64, S=1024, D=64) bf16,
// with q pre-scaled by D^-0.5 = 0.125.
// ---------------------------------------------------------------------------
__global__ __launch_bounds__(256) void qkv_gemm_kernel(
    const float* __restrict__ x, const float* __restrict__ Win,
    const float* __restrict__ bin,
    unsigned short* __restrict__ qb, unsigned short* __restrict__ kb,
    unsigned short* __restrict__ vb) {
  __shared__ alignas(16) unsigned short As[128][32];
  __shared__ alignas(16) unsigned short Bs[64][32];

  const int m0 = blockIdx.y * 128;
  const int n0 = blockIdx.x * 64;
  const int tid = threadIdx.x;
  const int w = tid >> 5, lane = tid & 31, half = lane >> 4, ln = lane & 15;
  const int wm = (w >> 1) * 32, wn = (w & 1) * 32;

  v8f acc[2][2] = {};

  for (int kk = 0; kk < 32; ++kk) {
    const int k0 = kk * 32;
    __syncthreads();
    for (int i = tid; i < 128 * 32; i += 256) {
      int r = i >> 5, c = i & 31;
      As[r][c] = f2bf(x[(size_t)(m0 + r) * 1024 + k0 + c]);
    }
    for (int i = tid; i < 64 * 32; i += 256) {
      int r = i >> 5, c = i & 31;
      Bs[r][c] = f2bf(Win[(size_t)(n0 + r) * 1024 + k0 + c]);
    }
    __syncthreads();

    Frag a[2], b[2];
    for (int fa = 0; fa < 2; ++fa) {
      const unsigned short* ap = &As[wm + fa * 16 + ln][0];
      a[fa].q[0] = *(const u32x4*)(ap + half * 8);
      a[fa].q[1] = *(const u32x4*)(ap + 16 + half * 8);
    }
    for (int fb = 0; fb < 2; ++fb) {
      const unsigned short* bp = &Bs[wn + fb * 16 + ln][0];
      b[fb].q[0] = *(const u32x4*)(bp + half * 8);
      b[fb].q[1] = *(const u32x4*)(bp + 16 + half * 8);
    }
    for (int fa = 0; fa < 2; ++fa)
      for (int fb = 0; fb < 2; ++fb)
        acc[fa][fb] = __builtin_amdgcn_wmma_f32_16x16x32_bf16(
            false, a[fa].bf, false, b[fb].bf, (short)0, acc[fa][fb], false, false);
  }

  for (int fa = 0; fa < 2; ++fa)
    for (int fb = 0; fb < 2; ++fb) {
      const int n = n0 + wn + fb * 16 + ln;       // global output feature
      const float bias = bin[n];
      const int h = n / 192;
      const int cc = n - h * 192;
      const int which = cc >> 6;                  // 0=q 1=k 2=v
      const int d = cc & 63;
      unsigned short* dst = (which == 0) ? qb : (which == 1) ? kb : vb;
      const float scale = (which == 0) ? 0.125f : 1.0f;
      for (int r = 0; r < 8; ++r) {
        const int gm = m0 + wm + fa * 16 + half * 8 + r;  // row = s*B + b
        const int s = gm >> 2, bb = gm & 3;
        const int bh = bb * 16 + h;
        float val = (acc[fa][fb][r] + bias) * scale;
        dst[((size_t)(bh << 10) + s) * 64 + d] = f2bf(val);
      }
    }
}

// ---------------------------------------------------------------------------
// Kernel 2: attention for one (bh, 32-row q block).
//   aw = q k^T + bias, mask -> softmax_1 -> * abm -> p @ v
// ---------------------------------------------------------------------------
__global__ __launch_bounds__(256) void attn_kernel(
    const unsigned short* __restrict__ qb, const unsigned short* __restrict__ kb,
    const unsigned short* __restrict__ vb,
    const float* __restrict__ bias, const float* __restrict__ bias_mul,
    const unsigned char* __restrict__ kpm,
    unsigned short* __restrict__ attn_buf) {
  __shared__ alignas(16) float aw[32][1024];            // 128 KB, reused as bf16 p
  __shared__ alignas(16) unsigned short vt[64][128];    // 16 KB, v^T chunk

  const int q0 = blockIdx.x * 32;
  const int bh = blockIdx.y;
  const int bb = bh >> 4;
  const int tid = threadIdx.x;
  const int w = tid >> 5, lane = tid & 31, half = lane >> 4, ln = lane & 15;
  const size_t bhSS = (size_t)bh * 1024 * 1024;

  // ---- Phase 1: aw = q k^T + bias (mask -> -1e30), 8 waves x 128 keys ----
  Frag afr[2][2];
  for (int fa = 0; fa < 2; ++fa)
    for (int ks = 0; ks < 2; ++ks) {
      const unsigned short* ap =
          qb + ((size_t)(bh << 10) + q0 + fa * 16 + ln) * 64 + ks * 32;
      afr[fa][ks].q[0] = *(const u32x4*)(ap + half * 8);
      afr[fa][ks].q[1] = *(const u32x4*)(ap + 16 + half * 8);
    }

  for (int kt = 0; kt < 8; ++kt) {
    const int keyc = w * 128 + kt * 16 + ln;     // this lane's key column
    const unsigned short* kp = kb + ((size_t)(bh << 10) + keyc) * 64;
    v8f acc[2] = {};
    for (int ks = 0; ks < 2; ++ks) {
      Frag bfr;
      bfr.q[0] = *(const u32x4*)(kp + ks * 32 + half * 8);
      bfr.q[1] = *(const u32x4*)(kp + ks * 32 + 16 + half * 8);
      for (int fa = 0; fa < 2; ++fa)
        acc[fa] = __builtin_amdgcn_wmma_f32_16x16x32_bf16(
            false, afr[fa][ks].bf, false, bfr.bf, (short)0, acc[fa], false, false);
    }
    const bool masked = kpm[bb * 1024 + keyc] != 0;
    for (int fa = 0; fa < 2; ++fa)
      for (int r = 0; r < 8; ++r) {
        const int m = fa * 16 + half * 8 + r;
        float val = acc[fa][r] + bias[bhSS + (size_t)(q0 + m) * 1024 + keyc];
        if (masked) val = -1e30f;
        aw[m][keyc] = val;
      }
  }
  __syncthreads();

  // ---- Phase 2: softmax_1 + *abm, one wave per row, bf16 p in place ----
  for (int i = 0; i < 4; ++i) {
    const int m = w + 8 * i;
    float* row = &aw[m][0];
    float vmax = -3.4e38f;
    for (int j = lane; j < 1024; j += 32) vmax = fmaxf(vmax, row[j]);
    for (int off = 16; off > 0; off >>= 1)
      vmax = fmaxf(vmax, __shfl_xor(vmax, off, 32));
    float ssum = 0.f;
    for (int j = lane; j < 1024; j += 32) ssum += __expf(row[j] - vmax);
    for (int off = 16; off > 0; off >>= 1) ssum += __shfl_xor(ssum, off, 32);
    const float inv = 1.f / (__expf(-vmax) + ssum);
    unsigned short* prow = (unsigned short*)row;
    const float* abm = bias_mul + bhSS + (size_t)(q0 + m) * 1024;
    const unsigned char* mrow = kpm + bb * 1024;
    for (int j = lane; j < 1024; j += 32) {
      float e = __expf(row[j] - vmax) * inv;        // read f32 ...
      float ab = mrow[j] ? 0.f : abm[j];
      prow[j] = f2bf(e * ab);                       // ... then overwrite as bf16
    }
  }
  __syncthreads();

  // ---- Phase 3: attn = p @ v, 128-key chunks staged transposed in LDS ----
  const int qt = w >> 2, dt = w & 3;               // wave -> 16x16 output tile
  v8f oacc = {};
  for (int c = 0; c < 8; ++c) {
    for (int i = tid; i < 8192; i += 256) {
      const int d = i & 63, kl = i >> 6;
      vt[d][kl] = vb[((size_t)(bh << 10) + c * 128 + kl) * 64 + d];
    }
    __syncthreads();
    const unsigned short* prow = (const unsigned short*)&aw[qt * 16 + ln][0];
    for (int ks = 0; ks < 4; ++ks) {
      Frag a, b;
      const int kbase = c * 128 + ks * 32;
      a.q[0] = *(const u32x4*)(prow + kbase + half * 8);
      a.q[1] = *(const u32x4*)(prow + kbase + 16 + half * 8);
      const unsigned short* vp = &vt[dt * 16 + ln][ks * 32];
      b.q[0] = *(const u32x4*)(vp + half * 8);
      b.q[1] = *(const u32x4*)(vp + 16 + half * 8);
      oacc = __builtin_amdgcn_wmma_f32_16x16x32_bf16(
          false, a.bf, false, b.bf, (short)0, oacc, false, false);
    }
    __syncthreads();
  }

  const int h = bh & 15;
  for (int r = 0; r < 8; ++r) {
    const int m = qt * 16 + half * 8 + r;
    const int s = q0 + m;
    const int d = dt * 16 + ln;
    attn_buf[(size_t)(s * 4 + bb) * 1024 + h * 64 + d] = f2bf(oacc[r]);
  }
}

// ---------------------------------------------------------------------------
// Kernel 3: out(4096 x 1024, f32) = attn(bf16) * W_out^T + b_out
// ---------------------------------------------------------------------------
__global__ __launch_bounds__(256) void out_gemm_kernel(
    const unsigned short* __restrict__ attn_buf, const float* __restrict__ Wout,
    const float* __restrict__ bout, float* __restrict__ out) {
  __shared__ alignas(16) unsigned short As[128][32];
  __shared__ alignas(16) unsigned short Bs[64][32];

  const int m0 = blockIdx.y * 128;
  const int n0 = blockIdx.x * 64;
  const int tid = threadIdx.x;
  const int w = tid >> 5, lane = tid & 31, half = lane >> 4, ln = lane & 15;
  const int wm = (w >> 1) * 32, wn = (w & 1) * 32;

  v8f acc[2][2] = {};

  for (int kk = 0; kk < 32; ++kk) {
    const int k0 = kk * 32;
    __syncthreads();
    for (int i = tid; i < 128 * 32; i += 256) {
      int r = i >> 5, c = i & 31;
      As[r][c] = attn_buf[(size_t)(m0 + r) * 1024 + k0 + c];
    }
    for (int i = tid; i < 64 * 32; i += 256) {
      int r = i >> 5, c = i & 31;
      Bs[r][c] = f2bf(Wout[(size_t)(n0 + r) * 1024 + k0 + c]);
    }
    __syncthreads();

    Frag a[2], b[2];
    for (int fa = 0; fa < 2; ++fa) {
      const unsigned short* ap = &As[wm + fa * 16 + ln][0];
      a[fa].q[0] = *(const u32x4*)(ap + half * 8);
      a[fa].q[1] = *(const u32x4*)(ap + 16 + half * 8);
    }
    for (int fb = 0; fb < 2; ++fb) {
      const unsigned short* bp = &Bs[wn + fb * 16 + ln][0];
      b[fb].q[0] = *(const u32x4*)(bp + half * 8);
      b[fb].q[1] = *(const u32x4*)(bp + 16 + half * 8);
    }
    for (int fa = 0; fa < 2; ++fa)
      for (int fb = 0; fb < 2; ++fb)
        acc[fa][fb] = __builtin_amdgcn_wmma_f32_16x16x32_bf16(
            false, a[fa].bf, false, b[fb].bf, (short)0, acc[fa][fb], false, false);
  }

  for (int fa = 0; fa < 2; ++fa)
    for (int fb = 0; fb < 2; ++fb) {
      const int n = n0 + wn + fb * 16 + ln;
      const float bias = bout[n];
      for (int r = 0; r < 8; ++r) {
        const int gm = m0 + wm + fa * 16 + half * 8 + r;
        out[(size_t)gm * 1024 + n] = acc[fa][fb][r] + bias;
      }
    }
}

// ---------------------------------------------------------------------------
extern "C" void kernel_launch(void* const* d_in, const int* in_sizes, int n_in,
                              void* d_out, int out_size, void* d_ws, size_t ws_size,
                              hipStream_t stream) {
  (void)in_sizes; (void)n_in; (void)out_size; (void)ws_size;
  const float* x             = (const float*)d_in[0];
  const float* attn_bias     = (const float*)d_in[1];
  const float* attn_bias_mul = (const float*)d_in[2];
  const unsigned char* kpm   = (const unsigned char*)d_in[3];
  const float* W_in          = (const float*)d_in[4];
  const float* b_in          = (const float*)d_in[5];
  const float* W_out         = (const float*)d_in[6];
  const float* b_out         = (const float*)d_in[7];

  const size_t QKV = (size_t)64 * 1024 * 64;   // 4 Mi bf16 elements each
  unsigned short* qb = (unsigned short*)d_ws;
  unsigned short* kb = qb + QKV;
  unsigned short* vb = kb + QKV;
  unsigned short* ab = vb + QKV;               // attn in (S,B,E) layout

  qkv_gemm_kernel<<<dim3(48, 32), 256, 0, stream>>>(x, W_in, b_in, qb, kb, vb);
  attn_kernel<<<dim3(32, 64), 256, 0, stream>>>(qb, kb, vb, attn_bias,
                                                attn_bias_mul, kpm, ab);
  out_gemm_kernel<<<dim3(16, 32), 256, 0, stream>>>(ab, W_out, b_out,
                                                    (float*)d_out);
}
